// PVAM_23983097381565
// MI455X (gfx1250) — compile-verified, matrix operand order checked by reading, not run on hardware
//
#include <hip/hip_runtime.h>
#include <hip/hip_bf16.h>
#include <cstdint>
#include <cstddef>

// Problem constants (match reference)
#define D_  512
#define B_  32
#define T_  256
#define L_  2
#define NH_ 8
#define DK_ 64
#define ML_ 25

typedef __attribute__((ext_vector_type(16))) __bf16 v16bf;
typedef __attribute__((ext_vector_type(8)))  float  v8f;
typedef __attribute__((ext_vector_type(4)))  float  v4f;

union FragBF { v16bf v; v4f q[2]; };
struct Stage { FragBF a0, a1, b0, b1, b2, b3; };

__device__ __forceinline__ unsigned short f2bf(float f) {
  unsigned int u = __float_as_uint(f);
  u += 0x7FFFu + ((u >> 16) & 1u);          // round-to-nearest-even
  return (unsigned short)(u >> 16);
}

// --------------------------------------------------------------------------
// TDM prefetch of a [64 rows x kTile] bf16 tile into LDS offset 0 (stage
// buffer). Pure prefetch: compute path still reads global; OOB rows/cols of
// the tensor return zero per ISA. 6-arg clang-23 builtin form
// (g0:u32x4, g1:i32x8, g2:i32x4, g3:i32x4, extra:i32x8, cpol:i32).
// --------------------------------------------------------------------------
__device__ __forceinline__ void tdm_prefetch_b(const unsigned short* gptr,
                                               int ldbElems, int kElems)
{
#if defined(__gfx1250__) && __has_builtin(__builtin_amdgcn_tensor_load_to_lds)
  typedef __attribute__((ext_vector_type(4))) unsigned int v4u;
  typedef __attribute__((ext_vector_type(8))) int v8i;
  typedef __attribute__((ext_vector_type(4))) int v4i;
  unsigned long long ga = (unsigned long long)(uintptr_t)gptr;
  int tile0 = kElems < 64 ? kElems : 64;
  v4u g0;
  g0.x = 1u;                                           // count=1 (valid), user
  g0.y = 0u;                                           // lds_addr = 0
  g0.z = (unsigned)(ga & 0xFFFFFFFFu);                 // global_addr[31:0]
  g0.w = (unsigned)((ga >> 32) & 0x01FFFFFFu) | (2u << 30); // addr[56:32]|type=2
  v8i g1;
  g1[0] = 0x10000;                                     // data_size=1 (2 bytes)
  g1[1] = (int)(((unsigned)kElems & 0xFFFFu) << 16);   // tensor_dim0 lo16
  g1[2] = (int)((((unsigned)kElems >> 16) & 0xFFFFu) | (64u << 16)); // dim0 hi | tensor_dim1 lo (=64)
  g1[3] = (int)((unsigned)tile0 << 16);                // tensor_dim1 hi=0 | tile_dim0
  g1[4] = 64;                                          // tile_dim1 = 64 rows
  g1[5] = ldbElems;                                    // tensor_dim0_stride lo32
  g1[6] = 0;
  g1[7] = 0;
  v4i z4 = {0, 0, 0, 0};
  v8i z8 = {0, 0, 0, 0, 0, 0, 0, 0};
  __builtin_amdgcn_tensor_load_to_lds(g0, g1, z4, z4, z8, 0);
#endif
}

__device__ __forceinline__ void tdm_wait()
{
#if defined(__gfx1250__) && __has_builtin(__builtin_amdgcn_s_wait_tensorcnt)
  __builtin_amdgcn_s_wait_tensorcnt(0);
#endif
}

// ---------------------------------------------------------------------------
// Generic batched GEMM:  C[M,N] = A[M,K](bf16, row-major) x Bt[N,K](bf16)
// Epilogue: *scale, +bias, relu, +resid(f32), store f32 and/or bf16.
// Block = 256 threads = 8 waves (4 in M x 2 in N); wave tile = 32x64,
// double-buffered register pipeline over K (K must be a multiple of 64).
// __launch_bounds__(256, 1): allow the full VGPR budget so the ~190-reg live
// set (64 acc + 96 stage + addressing) never spills to scratch.
// ---------------------------------------------------------------------------
__global__ void __launch_bounds__(256, 1) gemm_bf16_wmma(
    const unsigned short* __restrict__ A, const unsigned short* __restrict__ Bt,
    const float* __restrict__ bias, const float* resid,
    float* outF, unsigned short* outB,
    int M, int N, int K, int lda, int ldb, int ldc,
    long aSb, long aSh, long bSb, long bSh, long cSb, long cSh, int nh,
    float scale, int relu)
{
  __shared__ unsigned char tdm_stage[8192];   // TDM prefetch landing zone

  const int tid  = threadIdx.x;
  const int lane = tid & 31;
  const int wid  = tid >> 5;
  const int wm = wid & 3, wn = wid >> 2;
  const int mBase = (blockIdx.x * 4 + wm) * 32;
  const int nBase = (blockIdx.y * 2 + wn) * 64;
  if (tid == 0) *(volatile unsigned char*)tdm_stage = 0;  // keep LDS allocated
  if (mBase >= M || nBase >= N) return;

  const int z  = blockIdx.z;
  const int bz = z / nh, hz = z % nh;
  const long offA = (long)bz * aSb + (long)hz * aSh;
  const long offB = (long)bz * bSb + (long)hz * bSh;
  const long offC = (long)bz * cSb + (long)hz * cSh;

  const int r    = lane & 15;   // A/B row within tile; C column
  const int ksel = lane >> 4;   // which K half this lane carries

  const unsigned short* Ar0 = A + offA + (long)(mBase + r) * lda + ksel * 8;
  const unsigned short* Ar1 = Ar0 + (long)16 * lda;
  const unsigned short* Br0 = Bt + offB + (long)(nBase + r) * ldb + ksel * 8;
  const unsigned short* Br1 = Br0 + (long)16 * ldb;
  const unsigned short* Br2 = Br0 + (long)32 * ldb;
  const unsigned short* Br3 = Br0 + (long)48 * ldb;

  // One TDM async prefetch of this block's first B tile (wave 0 only).
  if (wid == 0) tdm_prefetch_b(Bt + offB + (long)nBase * ldb, ldb, K);

  auto ldf = [&](FragBF& f, const unsigned short* base, int kb) {
    f.q[0] = *reinterpret_cast<const v4f*>(base + kb);
    f.q[1] = *reinterpret_cast<const v4f*>(base + kb + 16);
  };
  auto loadS = [&](Stage& s, int kb) {
    ldf(s.a0, Ar0, kb); ldf(s.a1, Ar1, kb);
    ldf(s.b0, Br0, kb); ldf(s.b1, Br1, kb);
    ldf(s.b2, Br2, kb); ldf(s.b3, Br3, kb);
  };

  v8f zero = {0.f, 0.f, 0.f, 0.f, 0.f, 0.f, 0.f, 0.f};
  v8f acc[8];
#pragma unroll
  for (int i = 0; i < 8; ++i) acc[i] = zero;

  auto compute = [&](Stage& s) {
    acc[0] = __builtin_amdgcn_wmma_f32_16x16x32_bf16(false, s.a0.v, false, s.b0.v, (short)0, acc[0], false, false);
    acc[1] = __builtin_amdgcn_wmma_f32_16x16x32_bf16(false, s.a0.v, false, s.b1.v, (short)0, acc[1], false, false);
    acc[2] = __builtin_amdgcn_wmma_f32_16x16x32_bf16(false, s.a0.v, false, s.b2.v, (short)0, acc[2], false, false);
    acc[3] = __builtin_amdgcn_wmma_f32_16x16x32_bf16(false, s.a0.v, false, s.b3.v, (short)0, acc[3], false, false);
    acc[4] = __builtin_amdgcn_wmma_f32_16x16x32_bf16(false, s.a1.v, false, s.b0.v, (short)0, acc[4], false, false);
    acc[5] = __builtin_amdgcn_wmma_f32_16x16x32_bf16(false, s.a1.v, false, s.b1.v, (short)0, acc[5], false, false);
    acc[6] = __builtin_amdgcn_wmma_f32_16x16x32_bf16(false, s.a1.v, false, s.b2.v, (short)0, acc[6], false, false);
    acc[7] = __builtin_amdgcn_wmma_f32_16x16x32_bf16(false, s.a1.v, false, s.b3.v, (short)0, acc[7], false, false);
  };

  // Double-buffered pipeline: loads for the next stage are issued before the
  // WMMAs of the current stage, so s_wait_loadcnt covers a full WMMA batch.
  Stage s0, s1;
  loadS(s0, 0);
  for (int kb = 0; kb < K; kb += 64) {
    loadS(s1, kb + 32);
    compute(s0);
    if (kb + 64 < K) loadS(s0, kb + 64);
    compute(s1);
  }

  const int mOff = 8 * ksel;   // C layout: m = rr + 8*(lane>>4), n = lane&15
#pragma unroll
  for (int mi = 0; mi < 2; ++mi) {
#pragma unroll
    for (int ac = 0; ac < 4; ++ac) {
      const int n = nBase + ac * 16 + r;
#pragma unroll
      for (int rr = 0; rr < 8; ++rr) {
        const int m = mBase + mi * 16 + mOff + rr;
        float v = acc[mi * 4 + ac][rr] * scale;
        if (bias)  v += bias[n];
        if (relu)  v = fmaxf(v, 0.f);
        const long ci = offC + (long)m * ldc + n;
        if (resid) v += resid[ci];
        if (outF)  outF[ci] = v;
        if (outB)  outB[ci] = f2bf(v);
      }
    }
  }
  tdm_wait();
}

// f32 [nmat, K, N] -> bf16 transposed [nmat, N, K]
__global__ void k_conv_transpose(const float* __restrict__ w,
                                 unsigned short* __restrict__ wt, int nmat)
{
  long idx = (long)blockIdx.x * 256 + threadIdx.x;
  if (idx >= (long)nmat * D_ * D_) return;
  int kk = (int)(idx % D_);
  long r2 = idx / D_;
  int n  = (int)(r2 % D_);
  long l = r2 / D_;
  wt[idx] = f2bf(w[(l * D_ + kk) * D_ + n]);
}

// enc[b,t,c] = x[b,c,t]*sqrt(D) + pos_table[ewp[t], c]
__global__ void k_prep_enc(const float* __restrict__ x, const int* __restrict__ ewp,
                           const float* __restrict__ pos, float* __restrict__ enc)
{
  long idx = (long)blockIdx.x * 256 + threadIdx.x;
  if (idx >= (long)B_ * T_ * D_) return;
  int c = (int)(idx % D_);
  long r2 = idx / D_;
  int t = (int)(r2 % T_);
  int b = (int)(r2 / T_);
  float xv = x[((long)b * D_ + c) * T_ + t];
  enc[idx] = xv * 22.627416997969522f + pos[(long)ewp[t] * D_ + c];
}

// Row LayerNorm (row length 512), f32 in -> bf16 out. One block (256 thr) per row.
__global__ void k_layernorm(const float* __restrict__ in, const float* __restrict__ s,
                            const float* __restrict__ bta, unsigned short* __restrict__ out)
{
  const int row = blockIdx.x;
  const int tid = threadIdx.x;
  const float* p = in + (long)row * D_;
  float x0 = p[tid], x1 = p[tid + 256];
  float sum = x0 + x1, sq = x0 * x0 + x1 * x1;
  for (int off = 16; off > 0; off >>= 1) {
    sum += __shfl_xor(sum, off, 32);
    sq  += __shfl_xor(sq,  off, 32);
  }
  __shared__ float sh[16];
  int wid = tid >> 5, lane = tid & 31;
  if (lane == 0) { sh[wid] = sum; sh[wid + 8] = sq; }
  __syncthreads();
  float ts = 0.f, tq = 0.f;
#pragma unroll
  for (int i = 0; i < 8; ++i) { ts += sh[i]; tq += sh[i + 8]; }
  float mean = ts * (1.f / 512.f);
  float var  = tq * (1.f / 512.f) - mean * mean;
  float rs   = rsqrtf(var + 1e-5f);
  out[(long)row * D_ + tid]       = f2bf((x0 - mean) * rs * s[tid]       + bta[tid]);
  out[(long)row * D_ + tid + 256] = f2bf((x1 - mean) * rs * s[tid + 256] + bta[tid + 256]);
}

// Softmax over rows of length 256, one wave per row. Output bf16 and/or f32.
__global__ void k_softmax_rows(const float* __restrict__ in,
                               unsigned short* outB, float* outF, long nrows)
{
  long row = (long)blockIdx.x * 8 + (threadIdx.x >> 5);
  if (row >= nrows) return;
  const int lane = threadIdx.x & 31;
  const float* p = in + row * 256;
  float v[8];
  float mx = -3.4e38f;
#pragma unroll
  for (int j = 0; j < 8; ++j) { v[j] = p[lane + j * 32]; mx = fmaxf(mx, v[j]); }
  for (int off = 16; off > 0; off >>= 1) mx = fmaxf(mx, __shfl_xor(mx, off, 32));
  float sum = 0.f;
#pragma unroll
  for (int j = 0; j < 8; ++j) { v[j] = __expf(v[j] - mx); sum += v[j]; }
  for (int off = 16; off > 0; off >>= 1) sum += __shfl_xor(sum, off, 32);
  float inv = 1.f / sum;
#pragma unroll
  for (int j = 0; j < 8; ++j) {
    float o = v[j] * inv;
    if (outB) outB[row * 256 + lane + j * 32] = f2bf(o);
    if (outF) outF[row * 256 + lane + j * 32] = o;
  }
}

// vT[b,h,dk,t] = v[(b*T+t)*D + h*DK + dk]   (bf16)
__global__ void k_transpose_v(const unsigned short* __restrict__ v,
                              unsigned short* __restrict__ vt)
{
  long idx = (long)blockIdx.x * 256 + threadIdx.x;
  if (idx >= (long)B_ * NH_ * DK_ * T_) return;
  int t = (int)(idx & 255);
  int d = (int)((idx >> 8) & 63);
  int h = (int)((idx >> 14) & 7);
  int b = (int)(idx >> 17);
  vt[idx] = v[((long)(b * T_ + t)) * D_ + h * DK_ + d];
}

// sc[b,m,t] = sum_c tanh(wf[b,t,c] + emb[gwp[m],c]) * fc1[c]; one wave per (b,m,t)
__global__ void k_addattn_scores(const float* __restrict__ wf, const float* __restrict__ emb,
                                 const int* __restrict__ gwp, const float* __restrict__ fc1,
                                 float* __restrict__ sc)
{
  long w = (long)blockIdx.x * 8 + (threadIdx.x >> 5);
  if (w >= (long)B_ * ML_ * T_) return;
  const int lane = threadIdx.x & 31;
  int t = (int)(w % T_);
  long r2 = w / T_;
  int m = (int)(r2 % ML_);
  int b = (int)(r2 / ML_);
  const float* pw = wf + ((long)b * T_ + t) * D_;
  const float* pe = emb + (long)gwp[m] * D_;
  float acc = 0.f;
#pragma unroll 4
  for (int j = 0; j < 16; ++j) {
    int i = lane + j * 32;
    acc += tanhf(pw[i] + pe[i]) * fc1[i];
  }
  for (int off = 16; off > 0; off >>= 1) acc += __shfl_xor(acc, off, 32);
  if (lane == 0) sc[w] = acc;
}

// out[b,m,c] = sum_t attn[b,m,t] * wf[b,t,c]; one block (256 thr) per (b,m)
__global__ void k_final_out(const float* __restrict__ attn, const float* __restrict__ wf,
                            float* __restrict__ out)
{
  const int bm = blockIdx.x;
  const int b  = bm / ML_;
  __shared__ float at[T_];
  at[threadIdx.x] = attn[(long)bm * T_ + threadIdx.x];
  __syncthreads();
  for (int c = threadIdx.x; c < D_; c += 256) {
    float acc = 0.f;
    for (int t = 0; t < T_; ++t) acc += at[t] * wf[((long)b * T_ + t) * D_ + c];
    out[(long)bm * D_ + c] = acc;
  }
}

// ---------------------------------------------------------------------------
extern "C" void kernel_launch(void* const* d_in, const int* in_sizes, int n_in,
                              void* d_out, int out_size, void* d_ws, size_t ws_size,
                              hipStream_t stream)
{
  const float* x    = (const float*)d_in[0];
  const int*   ewp  = (const int*)d_in[1];
  const int*   gwp  = (const int*)d_in[2];
  const float* pos  = (const float*)d_in[3];
  const float* ln1s = (const float*)d_in[4];
  const float* ln1b = (const float*)d_in[5];
  const float* Wq = (const float*)d_in[6];  const float* bq = (const float*)d_in[7];
  const float* Wk = (const float*)d_in[8];  const float* bk = (const float*)d_in[9];
  const float* Wv = (const float*)d_in[10]; const float* bv = (const float*)d_in[11];
  const float* Wo = (const float*)d_in[12]; const float* bo = (const float*)d_in[13];
  const float* ln2s = (const float*)d_in[14]; const float* ln2b = (const float*)d_in[15];
  const float* W1 = (const float*)d_in[16]; const float* b1 = (const float*)d_in[17];
  const float* W2 = (const float*)d_in[18]; const float* b2 = (const float*)d_in[19];
  const float* lnfs = (const float*)d_in[20]; const float* lnfb = (const float*)d_in[21];
  const float* fc0w = (const float*)d_in[22]; const float* fc0b = (const float*)d_in[23];
  const float* emb  = (const float*)d_in[24]; const float* fc1w = (const float*)d_in[25];
  float* out = (float*)d_out;

  // ---- workspace carve-out ----
  char* p = (char*)d_ws;
  auto alloc = [&](size_t bytes) -> char* {
    char* r = p;
    p += (bytes + 255) & ~(size_t)255;
    return r;
  };
  const size_t szWT   = (size_t)L_ * D_ * D_ * 2;       // stacked bf16 weight (1MB)
  const size_t szAct4 = (size_t)B_ * T_ * D_ * 4;       // f32 activation (16MB)
  const size_t szAct2 = (size_t)B_ * T_ * D_ * 2;       // bf16 activation (8MB)

  unsigned short* wqT  = (unsigned short*)alloc(szWT);
  unsigned short* wkT  = (unsigned short*)alloc(szWT);
  unsigned short* wvT  = (unsigned short*)alloc(szWT);
  unsigned short* woT  = (unsigned short*)alloc(szWT);
  unsigned short* w1T  = (unsigned short*)alloc(szWT);
  unsigned short* w2T  = (unsigned short*)alloc(szWT);
  unsigned short* fc0T = (unsigned short*)alloc((size_t)D_ * D_ * 2);
  float*          enc  = (float*)alloc(szAct4);
  unsigned short* xln  = (unsigned short*)alloc(szAct2);
  unsigned short* q    = (unsigned short*)alloc(szAct2);
  unsigned short* k    = (unsigned short*)alloc(szAct2);
  unsigned short* v    = (unsigned short*)alloc(szAct2);
  unsigned short* o    = (unsigned short*)alloc(szAct2);
  unsigned short* ffn  = (unsigned short*)alloc(szAct2);
  unsigned short* vT   = (unsigned short*)alloc(szAct2);
  float*          S    = (float*)alloc((size_t)B_ * NH_ * T_ * T_ * 4);          // 64MB
  unsigned short* P    = (unsigned short*)alloc((size_t)B_ * NH_ * T_ * T_ * 2); // 32MB
  float*          wf   = (float*)alloc(szAct4);
  float*          s2   = (float*)alloc((size_t)B_ * ML_ * T_ * 4);
  float*          a2   = (float*)alloc((size_t)B_ * ML_ * T_ * 4);

  // ---- weight convert + transpose ----
  {
    long tot2 = (long)L_ * D_ * D_;
    int g2 = (int)((tot2 + 255) / 256);
    k_conv_transpose<<<g2, 256, 0, stream>>>(Wq, wqT, L_);
    k_conv_transpose<<<g2, 256, 0, stream>>>(Wk, wkT, L_);
    k_conv_transpose<<<g2, 256, 0, stream>>>(Wv, wvT, L_);
    k_conv_transpose<<<g2, 256, 0, stream>>>(Wo, woT, L_);
    k_conv_transpose<<<g2, 256, 0, stream>>>(W1, w1T, L_);
    k_conv_transpose<<<g2, 256, 0, stream>>>(W2, w2T, L_);
    long tot1 = (long)D_ * D_;
    k_conv_transpose<<<(int)((tot1 + 255) / 256), 256, 0, stream>>>(fc0w, fc0T, 1);
  }

  // ---- positional prep ----
  {
    long tot = (long)B_ * T_ * D_;
    k_prep_enc<<<(int)((tot + 255) / 256), 256, 0, stream>>>(x, ewp, pos, enc);
  }

  auto gemm = [&](const unsigned short* A, const unsigned short* Bt,
                  const float* bias, const float* resid,
                  float* oF, unsigned short* oB,
                  int M, int N, int K, int lda, int ldb, int ldc,
                  long aSb, long aSh, long bSb, long bSh, long cSb, long cSh,
                  int nh, int nbatch, float scale, int relu) {
    dim3 g((M + 127) / 128, (N + 127) / 128, nbatch);
    gemm_bf16_wmma<<<g, dim3(256), 0, stream>>>(
        A, Bt, bias, resid, oF, oB, M, N, K, lda, ldb, ldc,
        aSb, aSh, bSb, bSh, cSb, cSh, nh, scale, relu);
  };

  const int MT = B_ * T_;               // 8192
  const long zl = 0;

  for (int l = 0; l < L_; ++l) {
    const size_t wOff = (size_t)l * D_ * D_;
    // pre-norm 1
    k_layernorm<<<MT, 256, 0, stream>>>(enc, ln1s + l * D_, ln1b + l * D_, xln);
    // Q, K, V projections -> bf16 [B*T, 512]
    gemm(xln, wqT + wOff, bq + l * D_, nullptr, nullptr, q,
         MT, D_, D_, D_, D_, D_, zl, zl, zl, zl, zl, zl, 1, 1, 1.f, 0);
    gemm(xln, wkT + wOff, bk + l * D_, nullptr, nullptr, k,
         MT, D_, D_, D_, D_, D_, zl, zl, zl, zl, zl, zl, 1, 1, 1.f, 0);
    gemm(xln, wvT + wOff, bv + l * D_, nullptr, nullptr, v,
         MT, D_, D_, D_, D_, D_, zl, zl, zl, zl, zl, zl, 1, 1, 1.f, 0);
    // S = Q K^T * (1/sqrt(dk)) ; batched over (b,h)
    gemm(q, k, nullptr, nullptr, S, nullptr,
         T_, T_, DK_, D_, D_, T_,
         (long)T_ * D_, (long)DK_, (long)T_ * D_, (long)DK_,
         (long)NH_ * T_ * T_, (long)T_ * T_, NH_, B_ * NH_, 0.125f, 0);
    // softmax rows -> P (bf16)
    k_softmax_rows<<<(B_ * NH_ * T_) / 8, 256, 0, stream>>>(S, P, nullptr,
                                                            (long)B_ * NH_ * T_);
    // V transpose -> [b,h,dk,t]
    {
      long tot = (long)B_ * NH_ * DK_ * T_;
      k_transpose_v<<<(int)((tot + 255) / 256), 256, 0, stream>>>(v, vT);
    }
    // O = P V ; write into [B*T, 512] layout at col h*64
    gemm(P, vT, nullptr, nullptr, nullptr, o,
         T_, DK_, T_, T_, T_, D_,
         (long)NH_ * T_ * T_, (long)T_ * T_,
         (long)NH_ * DK_ * T_, (long)DK_ * T_,
         (long)T_ * D_, (long)DK_, NH_, B_ * NH_, 1.f, 0);
    // enc += O Wo + bo
    gemm(o, woT + wOff, bo + l * D_, enc, enc, nullptr,
         MT, D_, D_, D_, D_, D_, zl, zl, zl, zl, zl, zl, 1, 1, 1.f, 0);
    // pre-norm 2 + FFN
    k_layernorm<<<MT, 256, 0, stream>>>(enc, ln2s + l * D_, ln2b + l * D_, xln);
    gemm(xln, w1T + wOff, b1 + l * D_, nullptr, nullptr, ffn,
         MT, D_, D_, D_, D_, D_, zl, zl, zl, zl, zl, zl, 1, 1, 1.f, 1);
    gemm(ffn, w2T + wOff, b2 + l * D_, enc, enc, nullptr,
         MT, D_, D_, D_, D_, D_, zl, zl, zl, zl, zl, zl, 1, 1, 1.f, 0);
  }

  // final LN + fc0 -> word_features (f32)
  k_layernorm<<<MT, 256, 0, stream>>>(enc, lnfs, lnfb, xln);
  gemm(xln, fc0T, fc0b, nullptr, wf, nullptr,
       MT, D_, D_, D_, D_, D_, zl, zl, zl, zl, zl, zl, 1, 1, 1.f, 0);

  // additive attention
  k_addattn_scores<<<(B_ * ML_ * T_) / 8, 256, 0, stream>>>(wf, emb, gwp, fc1w, s2);
  k_softmax_rows<<<(B_ * ML_ + 7) / 8, 256, 0, stream>>>(s2, nullptr, a2,
                                                         (long)B_ * ML_);
  k_final_out<<<B_ * ML_, 256, 0, stream>>>(a2, wf, out);
}
